// EdgeAttention_89979564851989
// MI455X (gfx1250) — matrix operand order repaired and checked by reading, not compiled, not run on hardware
//
#include <hip/hip_runtime.h>
#include <hip/hip_bf16.h>
#include <math.h>

// ---------------------------------------------------------------------------
// EdgeAttention for MI455X (gfx1250, wave32, WMMA bf16 + TDM tile staging)
// ---------------------------------------------------------------------------

typedef __attribute__((ext_vector_type(16))) __bf16 v16bf;
typedef __attribute__((ext_vector_type(8)))  __bf16 v8bf;
typedef __attribute__((ext_vector_type(8)))  float  v8f;
typedef __attribute__((ext_vector_type(4)))  unsigned int u32x4;
typedef __attribute__((ext_vector_type(4)))  int    i32x4;
typedef __attribute__((ext_vector_type(8)))  int    i32x8;
typedef unsigned short bf16raw;   // host-safe handle for bf16 storage

#define D_V    256
#define D_E    128
#define D_ATTN 128

#if defined(__HIP_DEVICE_COMPILE__) && defined(__gfx1250__) && \
    __has_builtin(__builtin_amdgcn_tensor_load_to_lds) &&      \
    __has_builtin(__builtin_amdgcn_s_wait_tensorcnt)
#define USE_TDM 1
#else
#define USE_TDM 0
#endif

__device__ __forceinline__ float lrelu(float x) { return x >= 0.f ? x : 0.01f * x; }

__device__ __forceinline__ v16bf cat8(v8bf lo, v8bf hi) {
  return __builtin_shufflevector(lo, hi, 0,1,2,3,4,5,6,7,8,9,10,11,12,13,14,15);
}

// order-preserving float <-> uint mapping for atomicMax-based segment max
__device__ __forceinline__ unsigned flip_f32(float f) {
  unsigned u = __float_as_uint(f);
  return (u & 0x80000000u) ? ~u : (u | 0x80000000u);
}
__device__ __forceinline__ float unflip_f32(unsigned u) {
  unsigned v = (u & 0x80000000u) ? (u ^ 0x80000000u) : ~u;
  return __uint_as_float(v);
}

#if USE_TDM
// Issue a TDM DMA of a 2D f32 tile (tile_rows x cols) global -> LDS.
// Uniform arguments only; issued once per executing wave (EXEC is ignored).
__device__ __forceinline__ void tdm_load_tile_f32(
    const float* gsrc, void* lds_dst, unsigned tensor_rows, unsigned tile_rows,
    unsigned cols, unsigned row_stride_elems)
{
  const unsigned           lds_off = (unsigned)(unsigned long long)lds_dst;
  const unsigned long long ga      = (unsigned long long)gsrc;

  u32x4 g0;
  g0[0] = 1u;                                              // count=1 (valid D#)
  g0[1] = lds_off;                                         // lds_addr (bytes)
  g0[2] = (unsigned)(ga & 0xFFFFFFFFull);                  // global_addr[31:0]
  g0[3] = (unsigned)((ga >> 32) & 0x01FFFFFFull)           // global_addr[56:32]
        | (2u << 30);                                      // type = 2 ("image")

  i32x8 g1;
  g1[0] = (int)(2u << 16);                                 // data_size=2 (4B); mask=0
  g1[1] = (int)((cols & 0xFFFFu) << 16);                   // tensor_dim0[15:0]
  g1[2] = (int)(((cols >> 16) & 0xFFFFu)                   // tensor_dim0[31:16]
        | ((tensor_rows & 0xFFFFu) << 16));                // tensor_dim1[15:0]
  g1[3] = (int)(((tensor_rows >> 16) & 0xFFFFu)            // tensor_dim1[31:16]
        | ((cols & 0xFFFFu) << 16));                       // tile_dim0
  g1[4] = (int)(tile_rows & 0xFFFFu);                      // tile_dim1 (tile_dim2=0)
  g1[5] = (int)row_stride_elems;                           // tensor_dim0_stride[31:0]
  g1[6] = 0;                                               // stride0 hi / stride1 lo
  g1[7] = 0;

  const i32x4 z4 = {0, 0, 0, 0};
#if __clang_major__ >= 23
  const i32x8 z8 = {0, 0, 0, 0, 0, 0, 0, 0};
  __builtin_amdgcn_tensor_load_to_lds(g0, g1, z4, z4, z8, 0);
#else
  __builtin_amdgcn_tensor_load_to_lds(g0, g1, z4, z4, 0);
#endif
}
#endif // USE_TDM

// ---------------------------------------------------------------------------
// Kernel 1: init — zero out/denom/mkeys, convert weights f32 -> bf16
// ---------------------------------------------------------------------------
__global__ __launch_bounds__(256) void init_kernel(
    float* __restrict__ out, float* __restrict__ denom, unsigned* __restrict__ mkeys,
    const float* __restrict__ Wq, const float* __restrict__ Wk, const float* __restrict__ Wv,
    bf16raw* __restrict__ wq_bf, bf16raw* __restrict__ wk_bf, bf16raw* __restrict__ wv_bf,
    int N)
{
  const long idx = (long)blockIdx.x * 256 + threadIdx.x;
  const long outN = (long)N * D_ATTN;
  if (idx < outN) out[idx] = 0.f;
  if (idx < N) { denom[idx] = 0.f; mkeys[idx] = 0u; }
  if (idx < D_ATTN * D_V) {
    ((__bf16*)wq_bf)[idx] = (__bf16)Wq[idx];
    ((__bf16*)wk_bf)[idx] = (__bf16)Wk[idx];
  }
  if (idx < D_ATTN * D_E) {
    ((__bf16*)wv_bf)[idx] = (__bf16)Wv[idx];
  }
}

// ---------------------------------------------------------------------------
// Kernel 2: Q/K per-node GEMM.  Q = lrelu(nodes @ Wq^T + bq), same for K.
// Block = 256 threads (8 waves), handles 16 nodes x 128 outputs x {Q,K}.
// ---------------------------------------------------------------------------
__global__ __launch_bounds__(256) void qk_kernel(
    const float* __restrict__ nodes,
    const bf16raw* __restrict__ wq_bf, const float* __restrict__ bq,
    const bf16raw* __restrict__ wk_bf, const float* __restrict__ bk,
    float* __restrict__ Q, float* __restrict__ K, int N)
{
  __shared__ __bf16 sA[16][D_V];    // 16 node rows, bf16-converted (8 KB)

  const int  tid   = threadIdx.x;
  const long node0 = (long)blockIdx.x * 16;
  const bool full  = (node0 + 16 <= N);

  // cooperative stage: 16x256 f32 -> bf16 LDS (float4-vectorized)
  if (full) {
    for (int idx = tid; idx < 16 * (D_V / 4); idx += 256) {
      const int row = idx >> 6, c4 = idx & 63;
      const float4 v = ((const float4*)(nodes + (node0 + row) * D_V))[c4];
      __bf16* dst = &sA[row][c4 * 4];
      dst[0] = (__bf16)v.x; dst[1] = (__bf16)v.y;
      dst[2] = (__bf16)v.z; dst[3] = (__bf16)v.w;
    }
  } else {
    for (int idx = tid; idx < 16 * D_V; idx += 256) {
      const int row = idx >> 8, col = idx & (D_V - 1);
      const long nd = node0 + row;
      sA[row][col] = (__bf16)((nd < N) ? nodes[nd * D_V + col] : 0.f);
    }
  }
  __syncthreads();

  const int wave = tid >> 5, lane = tid & 31;
  const int half = lane >> 4, mn = lane & 15;

  for (int t = wave; t < 16; t += 8) {           // uniform per-wave: EXEC all-1 for WMMA
    const bool  isK   = (t >= 8);
    const int   ntile = t & 7;
    const __bf16* W    = (const __bf16*)(isK ? wk_bf : wq_bf);
    const float*  bias = isK ? bk : bq;
    float*        outp = isK ? K : Q;

    v8f c = {};
#pragma unroll
    for (int kb = 0; kb < D_V / 32; ++kb) {
      // A fragment (16x32 bf16): lane m = lane%16, k = i + 8*(i>=8) + 8*half
      v8bf alo = *(const v8bf*)&sA[mn][kb * 32 + 8 * half];
      v8bf ahi = *(const v8bf*)&sA[mn][kb * 32 + 16 + 8 * half];
      v16bf a = cat8(alo, ahi);
      // B fragment (32x16 bf16): B[k][n] = W[n][k]; k = i + 8*(i>=8) + 8*half, n = lane%16
      const __bf16* bp = W + (ntile * 16 + mn) * D_V + kb * 32;
      v16bf b = cat8(*(const v8bf*)(bp + 8 * half), *(const v8bf*)(bp + 16 + 8 * half));
      c = __builtin_amdgcn_wmma_f32_16x16x32_bf16(false, a, false, b, (short)0, c,
                                                  false, false);
    }

    const int   col = ntile * 16 + mn;
    const float bb  = bias[col];
    if (full) {
#pragma unroll
      for (int j = 0; j < 8; ++j) {              // C/D: M = j + 8*half, N = lane%16
        const int mm = j + 8 * half;
        outp[(node0 + mm) * D_ATTN + col] = lrelu(c[j] + bb);
      }
    } else {
#pragma unroll
      for (int j = 0; j < 8; ++j) {
        const int  mm = j + 8 * half;
        const long nd = node0 + mm;
        if (nd < N) outp[nd * D_ATTN + col] = lrelu(c[j] + bb);
      }
    }
  }
}

// ---------------------------------------------------------------------------
// Kernel 3: per-edge logits + segment max (receiver).  One wave per edge.
// Q,K (51 MB total) are L2-resident on a 192 MB L2.
// ---------------------------------------------------------------------------
__global__ __launch_bounds__(256) void logits_kernel(
    const float* __restrict__ Q, const float* __restrict__ K,
    const int* __restrict__ eidx, float* __restrict__ logits,
    unsigned* __restrict__ mkeys, int E)
{
  const int wave = threadIdx.x >> 5, lane = threadIdx.x & 31;
  const long e = (long)blockIdx.x * 8 + wave;
  if (e >= E) return;
  const int s = eidx[e];
  const int r = eidx[(long)E + e];

  const float4 qv = ((const float4*)(Q + (long)r * D_ATTN))[lane];
  const float4 kv = ((const float4*)(K + (long)s * D_ATTN))[lane];
  float part = qv.x * kv.x + qv.y * kv.y + qv.z * kv.z + qv.w * kv.w;
#pragma unroll
  for (int m = 16; m >= 1; m >>= 1) part += __shfl_xor(part, m, 32);

  const float logit = part * 0.08838834764831845f;   // 1/sqrt(128)
  if (lane == 0) {
    logits[e] = logit;
    atomicMax(&mkeys[r], flip_f32(logit));           // global_atomic_max_u32
  }
}

// ---------------------------------------------------------------------------
// Kernel 4: p = exp(logit - m[r]); denom[r] += p  (in-place over logits)
// ---------------------------------------------------------------------------
__global__ __launch_bounds__(256) void softmax_p_kernel(
    float* __restrict__ logits, const int* __restrict__ eidx,
    const unsigned* __restrict__ mkeys, float* __restrict__ denom, int E)
{
  const long e = (long)blockIdx.x * 256 + threadIdx.x;
  if (e >= E) return;
  const int   r = eidx[(long)E + e];
  const float m = unflip_f32(mkeys[r]);
  const float p = __expf(logits[e] - m);
  logits[e] = p;
  __hip_atomic_fetch_add(&denom[r], p, __ATOMIC_RELAXED, __HIP_MEMORY_SCOPE_AGENT);
}

// ---------------------------------------------------------------------------
// Kernel 5: fused V-GEMM + softmax-weighted scatter.
// Per block: 16 edges. The 16x128 f32 edge tile is DMA'd into LDS by the
// Tensor Data Mover (wave 0 issues; TENSORcnt waited), converted to bf16
// cooperatively, then v = lrelu(tile @ Wv^T + bv) via WMMA (4 K-steps,
// 8 waves = 8 column tiles); out[r[e]] += (p/denom)*v via f32 atomics.
// ---------------------------------------------------------------------------
__global__ __launch_bounds__(256) void v_scatter_kernel(
    const float* __restrict__ edges, const bf16raw* __restrict__ wv_bf,
    const float* __restrict__ bv, const int* __restrict__ eidx,
    const float* __restrict__ p, const float* __restrict__ denom,
    float* __restrict__ out, int E)
{
  __shared__ __bf16 sEbf[16][D_E];   // 4 KB bf16 tile for WMMA A fragments
  __shared__ float  sAttn[16];
  __shared__ int    sR[16];

  const int  tid  = threadIdx.x;
  const long e0   = (long)blockIdx.x * 16;
  const bool full = (e0 + 16 <= E);

#if USE_TDM
  __shared__ float sEf32[16][D_E];   // 8 KB TDM landing buffer

  if (tid < 32) {                    // wave 0 issues the DMA (EXEC ignored)
    const unsigned trows = (unsigned)(full ? 16 : (E - e0));
    tdm_load_tile_f32(edges + e0 * D_E, &sEf32[0][0], trows, 16u, D_E, D_E);
    __builtin_amdgcn_s_wait_tensorcnt(0);
  }
  if (tid >= 32 && tid < 48) {       // wave 1 precomputes attention coeffs
    const int  t = tid - 32;
    const long e = e0 + t;
    if (e < E) {
      const int r = eidx[(long)E + e];
      sR[t] = r;
      sAttn[t] = p[e] / denom[r];
    } else { sR[t] = 0; sAttn[t] = 0.f; }
  }
  __syncthreads();                   // DMA complete + attn coeffs visible

  // cooperative f32 -> bf16 conversion in LDS
  for (int idx = tid; idx < 16 * (D_E / 4); idx += 256) {
    const int row = idx >> 5, c4 = idx & 31;
    const float4 v = ((const float4*)&sEf32[row][0])[c4];
    __bf16* dst = &sEbf[row][c4 * 4];
    dst[0] = (__bf16)v.x; dst[1] = (__bf16)v.y;
    dst[2] = (__bf16)v.z; dst[3] = (__bf16)v.w;
  }
  __syncthreads();
#else
  if (tid == 0 && e0 + 16 < E)       // speculative prefetch of next tile
    __builtin_prefetch(&edges[(e0 + 16) * D_E], 0, 1);

  if (full) {
    for (int idx = tid; idx < 16 * (D_E / 4); idx += 256) {
      const int row = idx >> 5, c4 = idx & 31;
      const float4 v = ((const float4*)(edges + (e0 + row) * D_E))[c4];
      __bf16* dst = &sEbf[row][c4 * 4];
      dst[0] = (__bf16)v.x; dst[1] = (__bf16)v.y;
      dst[2] = (__bf16)v.z; dst[3] = (__bf16)v.w;
    }
  } else {
    for (int idx = tid; idx < 16 * D_E; idx += 256) {
      const int row = idx >> 7, col = idx & (D_E - 1);
      const long e = e0 + row;
      sEbf[row][col] = (__bf16)((e < E) ? edges[e * D_E + col] : 0.f);
    }
  }
  if (tid < 16) {
    const long e = e0 + tid;
    if (e < E) {
      const int r = eidx[(long)E + e];
      sR[tid] = r;
      sAttn[tid] = p[e] / denom[r];
    } else { sR[tid] = 0; sAttn[tid] = 0.f; }
  }
  __syncthreads();
#endif

  const int wave = tid >> 5, lane = tid & 31;
  const int half = lane >> 4, mn = lane & 15;
  const int ntile = wave;                               // 8 waves -> 8 col tiles
  const __bf16* W = (const __bf16*)wv_bf;

  v8f c = {};
#pragma unroll
  for (int kb = 0; kb < D_E / 32; ++kb) {
    v8bf alo = *(const v8bf*)&sEbf[mn][kb * 32 + 8 * half];
    v8bf ahi = *(const v8bf*)&sEbf[mn][kb * 32 + 16 + 8 * half];
    v16bf a = cat8(alo, ahi);
    const __bf16* bp = W + (ntile * 16 + mn) * D_E + kb * 32;
    v16bf b = cat8(*(const v8bf*)(bp + 8 * half), *(const v8bf*)(bp + 16 + 8 * half));
    c = __builtin_amdgcn_wmma_f32_16x16x32_bf16(false, a, false, b, (short)0, c,
                                                false, false);
  }

  const int   col = ntile * 16 + mn;
  const float bb  = bv[col];
  if (full) {
#pragma unroll
    for (int j = 0; j < 8; ++j) {
      const int   mm      = j + 8 * half;
      const float contrib = sAttn[mm] * lrelu(c[j] + bb);
      __hip_atomic_fetch_add(&out[(long)sR[mm] * D_ATTN + col], contrib,
                             __ATOMIC_RELAXED, __HIP_MEMORY_SCOPE_AGENT);
    }
  } else {
#pragma unroll
    for (int j = 0; j < 8; ++j) {
      const int  mm = j + 8 * half;
      const long e  = e0 + mm;
      if (e < E) {
        const float contrib = sAttn[mm] * lrelu(c[j] + bb);
        __hip_atomic_fetch_add(&out[(long)sR[mm] * D_ATTN + col], contrib,
                               __ATOMIC_RELAXED, __HIP_MEMORY_SCOPE_AGENT);
      }
    }
  }
}

// ---------------------------------------------------------------------------
// Host launcher
// ---------------------------------------------------------------------------
extern "C" void kernel_launch(void* const* d_in, const int* in_sizes, int n_in,
                              void* d_out, int out_size, void* d_ws, size_t ws_size,
                              hipStream_t stream)
{
  (void)n_in; (void)out_size; (void)ws_size;

  const float* nodes = (const float*)d_in[0];
  const float* edges = (const float*)d_in[1];
  const int*   eidx  = (const int*)  d_in[2];
  const float* Wq    = (const float*)d_in[3];
  const float* bq    = (const float*)d_in[4];
  const float* Wk    = (const float*)d_in[5];
  const float* bk    = (const float*)d_in[6];
  const float* Wv    = (const float*)d_in[7];
  const float* bv    = (const float*)d_in[8];

  const int N = in_sizes[0] / D_V;   // 50000
  const int E = in_sizes[1] / D_E;   // 1600000

  // workspace layout (256B aligned): Q, K, p, mkeys, denom, bf16 weights (~58 MB)
  char*  ws  = (char*)d_ws;
  size_t off = 0;
  auto alloc = [&](size_t bytes) -> void* {
    void* ptr = ws + off;
    off = (off + bytes + 255) & ~(size_t)255;
    return ptr;
  };
  float*    Q     = (float*)   alloc((size_t)N * D_ATTN * sizeof(float));
  float*    Kmat  = (float*)   alloc((size_t)N * D_ATTN * sizeof(float));
  float*    P     = (float*)   alloc((size_t)E * sizeof(float));
  unsigned* mkeys = (unsigned*)alloc((size_t)N * sizeof(unsigned));
  float*    denom = (float*)   alloc((size_t)N * sizeof(float));
  bf16raw*  wq_bf = (bf16raw*) alloc((size_t)D_ATTN * D_V * sizeof(bf16raw));
  bf16raw*  wk_bf = (bf16raw*) alloc((size_t)D_ATTN * D_V * sizeof(bf16raw));
  bf16raw*  wv_bf = (bf16raw*) alloc((size_t)D_ATTN * D_E * sizeof(bf16raw));

  float* out = (float*)d_out;

  const long outN = (long)N * D_ATTN;
  const int initBlocks = (int)((outN + 255) / 256);
  init_kernel<<<initBlocks, 256, 0, stream>>>(out, denom, mkeys, Wq, Wk, Wv,
                                              wq_bf, wk_bf, wv_bf, N);

  const int qkBlocks = (N + 15) / 16;
  qk_kernel<<<qkBlocks, 256, 0, stream>>>(nodes, wq_bf, bq, wk_bf, bk, Q, Kmat, N);

  const int lgBlocks = (E + 7) / 8;
  logits_kernel<<<lgBlocks, 256, 0, stream>>>(Q, Kmat, eidx, P, mkeys, E);

  const int smBlocks = (E + 255) / 256;
  softmax_p_kernel<<<smBlocks, 256, 0, stream>>>(P, eidx, mkeys, denom, E);

  const int vsBlocks = (E + 15) / 16;
  v_scatter_kernel<<<vsBlocks, 256, 0, stream>>>(edges, wv_bf, bv, eidx, P, denom,
                                                 out, E);
}
